// MultiRobotNetwork_56590489092479
// MI455X (gfx1250) — compile-verified, matrix operand order checked by reading, not run on hardware
//
#include <hip/hip_runtime.h>
#include <stdint.h>

// ============================ MI455X / CDNA5 =================================
// ~200 GFLOP of GEMM-shaped work vs ~8MB of real input traffic -> purely
// matrix-pipe bound.  All GEMM-shaped ops (convs via implicit im2col, MLPs,
// attention QKV / scores / PV) run on the wave32 WMMA pipe
// (v_wmma_f32_16x16x32_bf16, f32 accumulate).  Weights are packed once per
// launch into the ISA fragment layout (CDNA5 ISA 7.12.2).  Each 8-wave
// workgroup stages its packed-B K-panel into LDS (GLOBAL_LOAD_ASYNC_TO_LDS,
// ASYNCcnt-tracked) and 8 M-tiles share it -> 8x weight-traffic reduction,
// B fragments via ds_load_b128.  Workspace: ~295 MiB (layout below).
// =============================================================================

typedef __bf16 bf16_t;
typedef bf16_t bf16x16 __attribute__((ext_vector_type(16)));
typedef float  f32x8   __attribute__((ext_vector_type(8)));
typedef int    v4i_g   __attribute__((vector_size(16)));   // matches builtin V4i
typedef unsigned short u16;

#define DEV __device__ __forceinline__

#if defined(__has_builtin)
#if __has_builtin(__builtin_amdgcn_global_load_async_to_lds_b128) && \
    __has_builtin(__builtin_amdgcn_s_wait_asynccnt)
#define HAVE_ASYNC_LDS 1
#endif
#endif

DEV u16 f2bf(float f) {                       // round-to-nearest-even f32->bf16
  unsigned u = __float_as_uint(f);
  u += 0x7FFFu + ((u >> 16) & 1u);
  return (u16)(u >> 16);
}
DEV float bf2f(u16 h) { return __uint_as_float(((unsigned)h) << 16); }

union Frag { bf16x16 v; uint4 q[2]; u16 e[16]; };
union Acc  { f32x8 v; float f[8]; };

// A fragment from row-major bf16 with contiguous K: p = row base + kslab + off.
// Per ISA 7.12.2 (16-bit A 16x32): elems 0..7 hold K = off..off+7, elems 8..15
// hold K = off+16..off+23, off = (lane&16)?8:0  ->  two 16B loads.
DEV void load_ab(Frag& a, const u16* p) {
  a.q[0] = *(const uint4*)(p);
  a.q[1] = *(const uint4*)(p + 16);
}
DEV void load_pk(Frag& b, const u16* p) {     // packed: 16 contiguous u16
  b.q[0] = *(const uint4*)(p);
  b.q[1] = *(const uint4*)(p + 8);
}
DEV void zero_frag(Frag& a) { a.q[0] = make_uint4(0,0,0,0); a.q[1] = a.q[0]; }

DEV f32x8 wmma_bf16(const Frag& a, const Frag& b, f32x8 c) {
  return __builtin_amdgcn_wmma_f32_16x16x32_bf16(false, a.v, false, b.v,
                                                 (short)0, c, false, false);
}

// 16B global -> LDS copy, async (gfx1250 GLOBAL_LOAD_ASYNC_TO_LDS_B128) if
// available, else synchronous load + ds_store.
DEV void cp_g2l_16B(u16* l, const u16* g) {
#ifdef HAVE_ASYNC_LDS
  __builtin_amdgcn_global_load_async_to_lds_b128(
      (__attribute__((address_space(1))) v4i_g*)(u16*)g,
      (__attribute__((address_space(3))) v4i_g*)l, 0, 0);
#else
  *(uint4*)l = *(const uint4*)g;
#endif
}
DEV void cp_g2l_wait() {
#ifdef HAVE_ASYNC_LDS
  __builtin_amdgcn_s_wait_asynccnt(0);
#endif
}

// ---------------------------------------------------------------------------
// Weight packing: W (K x N, f32, row-major) -> bf16 fragments, layout
// out[((tn*ktiles+tk)*32 + lane)*16 + e], k = tk*32 + kmap(e,lane), n = tn*16+lane%16
// ---------------------------------------------------------------------------
__global__ void k_pack(const float* __restrict__ W, u16* __restrict__ out,
                       int N, int ktiles, int total) {
  int p = blockIdx.x * blockDim.x + threadIdx.x;
  if (p >= total) return;
  int e    = p & 15;
  int lane = (p >> 4) & 31;
  int tkn  = p >> 9;
  int tk = tkn % ktiles, tn = tkn / ktiles;
  int k = tk * 32 + (e < 8 ? e : e + 8) + ((lane & 16) ? 8 : 0);
  int n = tn * 16 + (lane & 15);
  out[p] = f2bf(W[(size_t)k * N + n]);
}

// ---------------------------------------------------------------------------
// WMMA GEMM:  Y(MxN) = act( X(MxK) * Wpacked + bias )
// One workgroup = 8 waves = 8 consecutive M-tiles sharing one N-tile; the
// packed-B K-panel (ktiles*1KB, K<=256) is staged to LDS once (async) and
// read back as ds_load_b128 fragments.  flags: bit0 relu, bit1 transposed
// store (Y[n*ldy+m], ldy=M).  Requires M%256==0, K%32==0, N%16==0.
// ---------------------------------------------------------------------------
__global__ __launch_bounds__(256) void k_gemm(
    const u16* __restrict__ X, const u16* __restrict__ Wp,
    const float* __restrict__ bias, u16* __restrict__ Y,
    int M, int K, int N, int ldx, int ldy, int flags) {
  __shared__ u16 ldsB[8 * 512];                 // up to K=256
  const int ntN = N >> 4, ktiles = K >> 5;
  const int tn = blockIdx.x % ntN;
  const int sm = blockIdx.x / ntN;              // M supertile (8 tiles)

  const u16* gB = Wp + (size_t)tn * ktiles * 512;
  const int chunks = ktiles * 64;               // 16B chunks in panel
  for (int c = threadIdx.x; c < chunks; c += 256)
    cp_g2l_16B(ldsB + c * 8, gB + c * 8);
  cp_g2l_wait();
  __syncthreads();

  const int lane = threadIdx.x & 31;
  const int wv = threadIdx.x >> 5;
  const int tm = sm * 8 + wv;
  const int ln = lane & 15;
  const int off = (lane & 16) ? 8 : 0;
  const int mbase = (lane & 16) >> 1;           // 0 or 8
  const u16* ap = X + (size_t)(tm * 16 + ln) * ldx + off;
  const u16* bp = ldsB + lane * 16;
  Acc acc; for (int i = 0; i < 8; i++) acc.f[i] = 0.f;
  for (int tk = 0; tk < ktiles; ++tk) {
    Frag a, b;
    load_ab(a, ap);
    load_pk(b, bp);                             // ds_load_b128 x2
    __builtin_prefetch(ap + 32, 0, 1);          // global_prefetch next A slab
    acc.v = wmma_bf16(a, b, acc.v);
    ap += 32; bp += 512;
  }
  const int n = tn * 16 + ln;
  const float bv = bias[n];
  const int m0 = tm * 16 + mbase;
  for (int i = 0; i < 8; ++i) {
    float v = acc.f[i] + bv;
    if (flags & 1) v = fmaxf(v, 0.f);
    if (flags & 2) Y[(size_t)n * ldy + (m0 + i)] = f2bf(v);
    else           Y[(size_t)(m0 + i) * ldy + n] = f2bf(v);
  }
}

// ---------------------------------------------------------------------------
// Implicit-im2col WMMA conv 3x3 SAME + bias + relu, NHWC, Cin%32==0, Cout%16==0.
// Same 8-M-tile / LDS-staged-B workgroup structure (panel up to 18KB).
// K slab tk covers neighbor nb = (tk*32)/Cin with channels cb..cb+31.
// ---------------------------------------------------------------------------
__global__ __launch_bounds__(256) void k_conv(
    const u16* __restrict__ A, const u16* __restrict__ Wp,
    const float* __restrict__ bias, u16* __restrict__ Y,
    int Mtot, int H, int W, int Cin, int Cout) {
  __shared__ u16 ldsB[18 * 512];                // up to K=576 (conv3)
  const int ntN = Cout >> 4, ktiles = (9 * Cin) >> 5;
  const int tn = blockIdx.x % ntN;
  const int sm = blockIdx.x / ntN;

  const u16* gB = Wp + (size_t)tn * ktiles * 512;
  const int chunks = ktiles * 64;
  for (int c = threadIdx.x; c < chunks; c += 256)
    cp_g2l_16B(ldsB + c * 8, gB + c * 8);
  cp_g2l_wait();
  __syncthreads();

  const int lane = threadIdx.x & 31;
  const int wv = threadIdx.x >> 5;
  const int tm = sm * 8 + wv;
  const int ln = lane & 15;
  const int off = (lane & 16) ? 8 : 0;
  const int mbase = (lane & 16) >> 1;
  const int m = tm * 16 + ln;
  const int x = m % W; int t = m / W; const int y = t % H; const int b = t / H;
  const u16* bp = ldsB + lane * 16;
  Acc acc; for (int i = 0; i < 8; i++) acc.f[i] = 0.f;
  for (int tk = 0; tk < ktiles; ++tk) {
    const int kb = tk * 32;
    const int nb = kb / Cin;
    const int cb = kb - nb * Cin;
    const int yy = y + nb / 3 - 1, xx = x + nb % 3 - 1;
    Frag a, bfr;
    if ((unsigned)yy < (unsigned)H && (unsigned)xx < (unsigned)W)
      load_ab(a, A + (((size_t)b * H + yy) * W + xx) * Cin + cb + off);
    else
      zero_frag(a);
    load_pk(bfr, bp);
    acc.v = wmma_bf16(a, bfr, acc.v);
    bp += 512;
  }
  const int n = tn * 16 + ln;
  const float bv = bias[n];
  const int m0 = tm * 16 + mbase;
  for (int i = 0; i < 8; ++i)
    Y[(size_t)(m0 + i) * Cout + n] = f2bf(fmaxf(acc.f[i] + bv, 0.f));
}

// ---------------------------------------------------------------------------
// Frontier self-attention (per (batch, 16-row query tile)), FD=128, NF=512.
// Pass1: S = scale*Q*K^T into LDS (waves split key tiles).  LDS softmax.
// Pass2: O = P*V (waves split output dims; V pre-transposed as Vt[d][b*512+key]),
// residual-added into F.  8 waves * 256 threads, 48KB LDS / workgroup.
// ---------------------------------------------------------------------------
__global__ __launch_bounds__(256) void k_fattn(
    const u16* __restrict__ Q, const u16* __restrict__ Km,
    const u16* __restrict__ Vt, u16* __restrict__ F) {
  __shared__ float S[16 * 512];
  __shared__ u16   P[16 * 512];
  __shared__ float red[256];
  const int tid = threadIdx.x;
  const int lane = tid & 31;
  const int wv = tid >> 5;
  const int b  = blockIdx.x >> 5;
  const int qt = blockIdx.x & 31;
  const int ln = lane & 15;
  const int off = (lane & 16) ? 8 : 0;
  const int mbase = (lane & 16) >> 1;
  const u16* qp = Q + ((size_t)b * 512 + qt * 16 + ln) * 128 + off;

  for (int kt = wv * 4; kt < wv * 4 + 4; ++kt) {
    Acc acc; for (int i = 0; i < 8; i++) acc.f[i] = 0.f;
    const u16* kp = Km + ((size_t)b * 512 + kt * 16 + ln) * 128 + off;
    for (int s = 0; s < 4; ++s) {            // FD = 4 slabs of 32
      Frag a, bb;
      load_ab(a, qp + s * 32);
      load_ab(bb, kp + s * 32);              // B col j = K row j (contiguous)
      acc.v = wmma_bf16(a, bb, acc.v);
    }
    for (int i = 0; i < 8; ++i)
      S[(mbase + i) * 512 + kt * 16 + ln] = acc.f[i] * 0.08838834764831845f;
  }
  __syncthreads();

  { // softmax over 512 keys; 16 threads per row, 32 cols per thread
    const int row = tid >> 4, seg = tid & 15;
    float* srow = &S[row * 512 + seg * 32];
    float mx = -3.4e38f;
    for (int i = 0; i < 32; i++) mx = fmaxf(mx, srow[i]);
    red[row * 16 + seg] = mx;
    __syncthreads();
    float rmax = red[row * 16];
    for (int i = 1; i < 16; i++) rmax = fmaxf(rmax, red[row * 16 + i]);
    __syncthreads();
    float sum = 0.f;
    for (int i = 0; i < 32; i++) { float e = __expf(srow[i] - rmax); srow[i] = e; sum += e; }
    red[row * 16 + seg] = sum;
    __syncthreads();
    float rsum = 0.f;
    for (int i = 0; i < 16; i++) rsum += red[row * 16 + i];
    const float inv = 1.f / rsum;
    u16* prow = &P[row * 512 + seg * 32];
    for (int i = 0; i < 32; i++) prow[i] = f2bf(srow[i] * inv);
  }
  __syncthreads();

  Acc oacc; for (int i = 0; i < 8; i++) oacc.f[i] = 0.f;
  const u16* prow = &P[(size_t)ln * 512 + off];
  const u16* vcol = Vt + (size_t)(wv * 16 + ln) * 131072 + (size_t)b * 512 + off;
  for (int kc = 0; kc < 16; ++kc) {          // 512 keys = 16 slabs of 32
    Frag a, bb;
    load_ab(a, prow + kc * 32);              // ds_load_b128 from LDS
    load_ab(bb, vcol + kc * 32);
    oacc.v = wmma_bf16(a, bb, oacc.v);
  }
  for (int i = 0; i < 8; ++i) {
    size_t fi = ((size_t)b * 512 + qt * 16 + mbase + i) * 128 + wv * 16 + ln;
    F[fi] = f2bf(bf2f(F[fi]) + oacc.f[i]);   // residual
  }
}

// --------------------------- scalar glue kernels ---------------------------
__global__ void k_conv1(const float* __restrict__ S, const float* __restrict__ W,
                        const float* __restrict__ B, u16* __restrict__ Y) {
  int idx = blockIdx.x * blockDim.x + threadIdx.x;
  if (idx >= 256 * 42 * 42 * 32) return;
  int c = idx & 31; int t = idx >> 5;
  int x = t % 42; t /= 42; int y = t % 42; int b = t / 42;
  float wl[9];
  for (int j = 0; j < 9; j++) wl[j] = W[j * 32 + c];
  const float bias = B[c];
  float best = 0.f;                           // relu-then-maxpool == max(0, max)
  for (int py = 0; py < 2; ++py)
    for (int px = 0; px < 2; ++px) {
      int yi = y * 2 + py, xi = x * 2 + px;
      float acc = bias;
      for (int dy = -1; dy <= 1; ++dy) {
        int yy = yi + dy; if ((unsigned)yy >= 84u) continue;
        for (int dx = -1; dx <= 1; ++dx) {
          int xx = xi + dx; if ((unsigned)xx >= 84u) continue;
          acc += S[((size_t)b * 84 + yy) * 84 + xx] * wl[(dy + 1) * 3 + (dx + 1)];
        }
      }
      best = fmaxf(best, acc);
    }
  Y[idx] = f2bf(best);
}

__global__ void k_pool(const u16* __restrict__ IN, u16* __restrict__ OUT) {
  int idx = blockIdx.x * blockDim.x + threadIdx.x;  // (256,21,21,64)
  if (idx >= 256 * 21 * 21 * 64) return;
  int c = idx & 63; int t = idx >> 6;
  int x = t % 21; t /= 21; int y = t % 21; int b = t / 21;
  const u16* p = IN + (((size_t)b * 42 + y * 2) * 42 + x * 2) * 64 + c;
  float v = fmaxf(fmaxf(bf2f(p[0]), bf2f(p[64])),
                  fmaxf(bf2f(p[42 * 64]), bf2f(p[42 * 64 + 64])));
  OUT[idx] = f2bf(v);
}

__global__ void k_gap(const u16* __restrict__ A, float* __restrict__ SF,
                      u16* __restrict__ SFB) {
  int idx = blockIdx.x * blockDim.x + threadIdx.x;  // (256,128)
  if (idx >= 256 * 128) return;
  int b = idx >> 7, c = idx & 127;
  const u16* p = A + (size_t)b * 441 * 128 + c;
  float s = 0.f;
  for (int i = 0; i < 441; i++) s += bf2f(p[(size_t)i * 128]);
  float m = s * (1.f / 441.f);
  SF[idx] = m; SFB[idx] = f2bf(m);
}

__global__ void k_mlp2(const float* __restrict__ X, const float* __restrict__ W,
                       const float* __restrict__ B, u16* __restrict__ Y, int M) {
  int idx = blockIdx.x * blockDim.x + threadIdx.x;  // (M,2)->(M,32) + relu
  if (idx >= M * 32) return;
  int n = idx & 31, r = idx >> 5;
  float v = B[n] + X[2 * r] * W[n] + X[2 * r + 1] * W[32 + n];
  Y[idx] = f2bf(fmaxf(v, 0.f));
}

__global__ void k_concat_rf(const u16* __restrict__ RF3, const u16* __restrict__ OC,
                            u16* __restrict__ RF) {
  int idx = blockIdx.x * blockDim.x + threadIdx.x;  // (512,128)
  if (idx >= 512 * 128) return;
  int c = idx & 127, r = idx >> 7, b = r >> 1;
  RF[idx] = (c < 64) ? RF3[r * 64 + c] : OC[b * 64 + (c - 64)];
}

__global__ void k_rattn(const u16* __restrict__ Q, const u16* __restrict__ Kt,
                        const u16* __restrict__ V, u16* __restrict__ RF) {
  int idx = blockIdx.x * blockDim.x + threadIdx.x;  // (b,r)
  if (idx >= 512) return;
  int b = idx >> 1;
  const u16* q = Q + (size_t)idx * 128;
  float s[2];
  for (int r = 0; r < 2; r++) {
    const u16* k = Kt + (size_t)(b * 2 + r) * 128;
    float d = 0.f;
    for (int c = 0; c < 128; c++) d += bf2f(q[c]) * bf2f(k[c]);
    s[r] = d * 0.08838834764831845f;
  }
  float mx = fmaxf(s[0], s[1]);
  float e0 = __expf(s[0] - mx), e1 = __expf(s[1] - mx);
  float inv = 1.f / (e0 + e1);
  float p0 = e0 * inv, p1 = e1 * inv;
  const u16* v0 = V + (size_t)(b * 2) * 128;
  const u16* v1 = v0 + 128;
  u16* rf = RF + (size_t)idx * 128;
  for (int c = 0; c < 128; c++)
    rf[c] = f2bf(bf2f(rf[c]) + p0 * bf2f(v0[c]) + p1 * bf2f(v1[c]));
}

__global__ void k_aff(const u16* __restrict__ Ro, const u16* __restrict__ Fo,
                      float* __restrict__ aff) {
  int idx = blockIdx.x * blockDim.x + threadIdx.x;  // (b,f)
  if (idx >= 256 * 512) return;
  int b = idx >> 9, f = idx & 511;
  const u16* fo = Fo + (size_t)(b * 512 + f) * 128;
  for (int r = 0; r < 2; r++) {
    const u16* ro = Ro + (size_t)(b * 2 + r) * 128;
    float d = 0.f;
    for (int c = 0; c < 128; c++) d += bf2f(ro[c]) * bf2f(fo[c]);
    aff[((size_t)b * 2 + r) * 512 + f] += d;
  }
}

__global__ void k_zero(float* p, int n) {
  int i = blockIdx.x * blockDim.x + threadIdx.x;
  if (i < n) p[i] = 0.f;
}

__global__ void k_softmax(const float* __restrict__ aff, float* __restrict__ out) {
  int lane = threadIdx.x & 31;                       // one wave per (b,r) row
  int wv = (blockIdx.x * blockDim.x + threadIdx.x) >> 5;
  if (wv >= 512) return;
  const float* row = aff + (size_t)wv * 512;
  const float s = 1.f / 3.f;
  float mx = -3.4e38f;
  for (int i = 0; i < 16; i++) mx = fmaxf(mx, row[lane + i * 32]);
  for (int o = 16; o; o >>= 1) mx = fmaxf(mx, __shfl_xor(mx, o, 32));
  mx *= s;
  float sum = 0.f;
  for (int i = 0; i < 16; i++) sum += __expf(row[lane + i * 32] * s - mx);
  for (int o = 16; o; o >>= 1) sum += __shfl_xor(sum, o, 32);
  float inv = 1.f / sum;
  float* orow = out + (size_t)wv * 512;
  for (int i = 0; i < 16; i++)
    orow[lane + i * 32] = __expf(row[lane + i * 32] * s - mx) * inv;
}

__global__ void k_vfeat(const float* __restrict__ SF, const u16* __restrict__ RF,
                        u16* __restrict__ VF) {
  int idx = blockIdx.x * blockDim.x + threadIdx.x;   // (256,256)
  if (idx >= 256 * 256) return;
  int b = idx >> 8, c = idx & 255;
  float v = (c < 128) ? SF[b * 128 + c]
                      : 0.5f * (bf2f(RF[(size_t)(b * 2) * 128 + c - 128]) +
                                bf2f(RF[(size_t)(b * 2 + 1) * 128 + c - 128]));
  VF[idx] = f2bf(v);
}

__global__ void k_vfinal(const u16* __restrict__ H, const float* __restrict__ W2,
                         const float* __restrict__ B2, float* __restrict__ out) {
  int b = blockIdx.x * blockDim.x + threadIdx.x;
  if (b >= 256) return;
  const u16* h = H + b * 64;
  float d = B2[0];
  for (int j = 0; j < 64; j++) d += bf2f(h[j]) * W2[j];
  out[b] = d;
}

// ------------------------------ workspace map ------------------------------
static const size_t PK_CONV2 = 0;
static const size_t PK_CONV3 = 36864;
static const size_t PK_RI2   = 184320;
static const size_t PK_RI3   = 188416;
static const size_t PK_FP2   = 196608;
static const size_t PK_FP3   = 200704;
static const size_t PK_OCC   = 217088;
static const size_t PK_RQ    = 233472;
static const size_t PK_RK    = 266240;
static const size_t PK_RV    = 299008;
static const size_t PK_FQ    = 331776;
static const size_t PK_FK    = 364544;
static const size_t PK_FV    = 397312;
static const size_t PK_OP    = 430080;
static const size_t PK_VH0   = 462848;
static const size_t PK_VH1   = 495616;
static const size_t SFEAT_F  = 1048576;          // (256,128) f32
static const size_t SFEAT_B  = 1179648;          // (256,128) bf16
static const size_t RF_O     = 1245184;          // (512,128) bf16
static const size_t RQ_O     = 1376256;
static const size_t RK_O     = 1507328;
static const size_t RV_O     = 1638400;
static const size_t RF1_O    = 1769472;          // (512,32)
static const size_t RF2_O    = 1802240;          // (512,64)
static const size_t RF3_O    = 1867776;          // (512,64)
static const size_t OCCV_O   = 1933312;          // (256,64)
static const size_t RO_O     = 1966080;          // (512,128)
static const size_t AFF_O    = 2097152;          // (256,2,512) f32
static const size_t VF_O     = 3145728;          // (256,256) bf16
static const size_t H0_O     = 3276800;          // (256,64)
static const size_t H1_O     = 3309568;
static const size_t ACT1_O   = (size_t)4   << 20;  // (256,42,42,32) bf16
static const size_t C2PRE_O  = (size_t)34  << 20;  // (256,42,42,64)
static const size_t ACT2_O   = (size_t)94  << 20;  // (256,21,21,64)
static const size_t ACT3_O   = (size_t)108 << 20;  // (256,21,21,128)
static const size_t FF1_O    = (size_t)138 << 20;  // (131072,32)
static const size_t FF2_O    = (size_t)147 << 20;  // (131072,64)
static const size_t F_O      = (size_t)164 << 20;  // (131072,128)
static const size_t FQ_O     = (size_t)197 << 20;
static const size_t FK_O     = (size_t)230 << 20;
static const size_t FVT_O    = (size_t)263 << 20;  // transposed (128,131072)
static const size_t FO_O     = FQ_O;               // alias: Fq dead before Fo

extern "C" void kernel_launch(void* const* d_in, const int* in_sizes, int n_in,
                              void* d_out, int out_size, void* d_ws, size_t ws_size,
                              hipStream_t stream) {
  (void)in_sizes; (void)n_in; (void)out_size; (void)ws_size;
  const float* states    = (const float*)d_in[0];
  const float* frontiers = (const float*)d_in[1];
  const float* rposes    = (const float*)d_in[2];
  const float* conv1_w = (const float*)d_in[4],  *conv1_b = (const float*)d_in[5];
  const float* conv2_w = (const float*)d_in[6],  *conv2_b = (const float*)d_in[7];
  const float* conv3_w = (const float*)d_in[8],  *conv3_b = (const float*)d_in[9];
  const float* ri1_w = (const float*)d_in[10], *ri1_b = (const float*)d_in[11];
  const float* ri2_w = (const float*)d_in[12], *ri2_b = (const float*)d_in[13];
  const float* ri3_w = (const float*)d_in[14], *ri3_b = (const float*)d_in[15];
  const float* fp1_w = (const float*)d_in[16], *fp1_b = (const float*)d_in[17];
  const float* fp2_w = (const float*)d_in[18], *fp2_b = (const float*)d_in[19];
  const float* fp3_w = (const float*)d_in[20], *fp3_b = (const float*)d_in[21];
  const float* occ_w = (const float*)d_in[22], *occ_b = (const float*)d_in[23];
  const float* rq_w  = (const float*)d_in[24], *rq_b  = (const float*)d_in[25];
  const float* rk_w  = (const float*)d_in[26], *rk_b  = (const float*)d_in[27];
  const float* rv_w  = (const float*)d_in[28], *rv_b  = (const float*)d_in[29];
  const float* fq_w  = (const float*)d_in[30], *fq_b  = (const float*)d_in[31];
  const float* fk_w  = (const float*)d_in[32], *fk_b  = (const float*)d_in[33];
  const float* fv_w  = (const float*)d_in[34], *fv_b  = (const float*)d_in[35];
  const float* op_w  = (const float*)d_in[36], *op_b  = (const float*)d_in[37];
  const float* vh0_1w = (const float*)d_in[38], *vh0_1b = (const float*)d_in[39];
  const float* vh0_2w = (const float*)d_in[40], *vh0_2b = (const float*)d_in[41];
  const float* vh1_1w = (const float*)d_in[42], *vh1_1b = (const float*)d_in[43];
  const float* vh1_2w = (const float*)d_in[44], *vh1_2b = (const float*)d_in[45];

  char* ws = (char*)d_ws;
  float* out = (float*)d_out;
  auto U = [&](size_t o) { return (u16*)(ws + o); };
  auto Ff = [&](size_t o) { return (float*)(ws + o); };

  auto pack = [&](const float* W, size_t off, int K, int N) {
    int total = (K / 32) * (N / 16) * 512;
    k_pack<<<(total + 255) / 256, 256, 0, stream>>>(W, U(off), N, K / 32, total);
  };
  auto gemm = [&](size_t xo, size_t pko, const float* bias, size_t yo,
                  int M, int K, int N, int ldx, int ldy, int flags) {
    int blocks = (M / 256) * (N / 16);          // 8 M-tiles per block
    k_gemm<<<blocks, 256, 0, stream>>>(U(xo), U(pko), bias, U(yo),
                                       M, K, N, ldx, ldy, flags);
  };

  // ---- pack all GEMM weights into WMMA fragment layout (once per call) ----
  pack(conv2_w, PK_CONV2, 288, 64);   pack(conv3_w, PK_CONV3, 576, 128);
  pack(ri2_w, PK_RI2, 32, 64);        pack(ri3_w, PK_RI3, 64, 64);
  pack(fp2_w, PK_FP2, 32, 64);        pack(fp3_w, PK_FP3, 64, 128);
  pack(occ_w, PK_OCC, 128, 64);
  pack(rq_w, PK_RQ, 128, 128); pack(rk_w, PK_RK, 128, 128); pack(rv_w, PK_RV, 128, 128);
  pack(fq_w, PK_FQ, 128, 128); pack(fk_w, PK_FK, 128, 128); pack(fv_w, PK_FV, 128, 128);
  pack(op_w, PK_OP, 128, 128);
  pack(vh0_1w, PK_VH0, 256, 64);      pack(vh1_1w, PK_VH1, 256, 64);

  // ---- CNN ----
  k_conv1<<<(256 * 42 * 42 * 32 + 255) / 256, 256, 0, stream>>>(states, conv1_w, conv1_b, U(ACT1_O));
  k_conv<<<(256 * 42 * 42 / 256) * 4, 256, 0, stream>>>(
      U(ACT1_O), U(PK_CONV2), conv2_b, U(C2PRE_O), 256 * 42 * 42, 42, 42, 32, 64);
  k_pool<<<(256 * 21 * 21 * 64 + 255) / 256, 256, 0, stream>>>(U(C2PRE_O), U(ACT2_O));
  k_conv<<<(256 * 21 * 21 / 256) * 8, 256, 0, stream>>>(
      U(ACT2_O), U(PK_CONV3), conv3_b, U(ACT3_O), 256 * 21 * 21, 21, 21, 64, 128);
  k_gap<<<(256 * 128 + 255) / 256, 256, 0, stream>>>(U(ACT3_O), Ff(SFEAT_F), U(SFEAT_B));

  // ---- robot init + occ + concat ----
  k_mlp2<<<(512 * 32 + 255) / 256, 256, 0, stream>>>(rposes, ri1_w, ri1_b, U(RF1_O), 512);
  gemm(RF1_O, PK_RI2, ri2_b, RF2_O, 512, 32, 64, 32, 64, 1);
  gemm(RF2_O, PK_RI3, ri3_b, RF3_O, 512, 64, 64, 64, 64, 0);
  gemm(SFEAT_B, PK_OCC, occ_b, OCCV_O, 256, 128, 64, 128, 64, 0);
  k_concat_rf<<<(512 * 128 + 255) / 256, 256, 0, stream>>>(U(RF3_O), U(OCCV_O), U(RF_O));

  // ---- frontier projection ----
  k_mlp2<<<(131072 * 32 + 255) / 256, 256, 0, stream>>>(frontiers, fp1_w, fp1_b, U(FF1_O), 131072);
  gemm(FF1_O, PK_FP2, fp2_b, FF2_O, 131072, 32, 64, 32, 64, 1);
  gemm(FF2_O, PK_FP3, fp3_b, F_O, 131072, 64, 128, 64, 128, 0);

  k_zero<<<(256 * 2 * 512 + 255) / 256, 256, 0, stream>>>(Ff(AFF_O), 256 * 2 * 512);

  // ---- GNN: 3 weight-shared layers ----
  for (int layer = 0; layer < 3; ++layer) {
    gemm(RF_O, PK_RQ, rq_b, RQ_O, 512, 128, 128, 128, 128, 0);
    gemm(RF_O, PK_RK, rk_b, RK_O, 512, 128, 128, 128, 128, 0);
    gemm(RF_O, PK_RV, rv_b, RV_O, 512, 128, 128, 128, 128, 0);
    k_rattn<<<2, 256, 0, stream>>>(U(RQ_O), U(RK_O), U(RV_O), U(RF_O));

    gemm(F_O, PK_FQ, fq_b, FQ_O, 131072, 128, 128, 128, 128, 0);
    gemm(F_O, PK_FK, fk_b, FK_O, 131072, 128, 128, 128, 128, 0);
    gemm(F_O, PK_FV, fv_b, FVT_O, 131072, 128, 128, 128, 131072, 2);  // transposed V
    k_fattn<<<256 * 32, 256, 0, stream>>>(U(FQ_O), U(FK_O), U(FVT_O), U(F_O));

    gemm(RF_O, PK_OP, op_b, RO_O, 512, 128, 128, 128, 128, 0);
    gemm(F_O, PK_OP, op_b, FO_O, 131072, 128, 128, 128, 128, 0);
    k_aff<<<(256 * 512 + 255) / 256, 256, 0, stream>>>(U(RO_O), U(FO_O), Ff(AFF_O));
  }

  // ---- outputs ----
  k_softmax<<<64, 256, 0, stream>>>(Ff(AFF_O), out);                 // (B,R,NF) probs
  k_vfeat<<<(256 * 256 + 255) / 256, 256, 0, stream>>>(Ff(SFEAT_F), U(RF_O), U(VF_O));
  gemm(VF_O, PK_VH0, vh0_1b, H0_O, 256, 256, 64, 256, 64, 1);
  gemm(VF_O, PK_VH1, vh1_1b, H1_O, 256, 256, 64, 256, 64, 1);
  k_vfinal<<<1, 256, 0, stream>>>(U(H0_O), vh0_2w, vh0_2b, out + 262144);
  k_vfinal<<<1, 256, 0, stream>>>(U(H1_O), vh1_2w, vh1_2b, out + 262144 + 256);
}
